// GNoMEModel_24945170055395
// MI455X (gfx1250) — compile-verified, matrix operand order checked by reading, not run on hardware
//
#include <hip/hip_runtime.h>
#include <cmath>

#define NN 16000
#define EE 128000
#define HD 128
#define ESTR 4   // edge strips (of 16 edges) per workgroup: amortizes B-fragment loads 4x

// Padded LDS row strides (in halfs / floats): +8 halfs (16B) keeps 16B
// alignment for b128 loads and staggers rows by 4 LDS banks, so the 16 rows
// read by an A-fragment load hit 16 distinct banks instead of 1.
#define MI_LD  392   // 384 + 8 halfs
#define HID_LD 136   // 128 + 8 halfs
#define AIN_LD 264   // 256 + 8 halfs
#define EMB_LD 40    // 32 + 8 halfs
#define PRE_LD 132   // 128 + 4 floats (4 dwords -> 4-bank stagger)

typedef __attribute__((ext_vector_type(16))) _Float16 v16h;
typedef __attribute__((ext_vector_type(8)))  float    v8f;

#define ZACC {0.f,0.f,0.f,0.f,0.f,0.f,0.f,0.f}

__device__ __forceinline__ v8f wmma16(v16h a, v16h b, v8f c) {
  // v_wmma_f32_16x16x32_f16: D = A*B + C, f32 accum
  return __builtin_amdgcn_wmma_f32_16x16x32_f16(false, a, false, b, (short)0, c, false, false);
}

// A-fragment loader from row-major f16 LDS tile with row stride ldk (halfs).
// Lane l (m = l&15, half = l>>4) holds K = {half*8..+7, 16+half*8..+7}
__device__ __forceinline__ v16h load_a(const _Float16* base, int ldk, int kbase, int lane) {
  int m = lane & 15, hf = lane >> 4;
  const _Float16* p = base + m * ldk + kbase + hf * 8;
  v16h a;
#pragma unroll
  for (int i = 0; i < 8; ++i) { a[i] = p[i]; a[8 + i] = p[16 + i]; }
  return a;
}

__device__ __forceinline__ float gelu_f(float x) {
  return 0.5f * x * (1.0f + erff(x * 0.70710678118654752440f));
}

// ---------------------------------------------------------------------------
// Pack fp32 weight (K x 128, row-major) into B-fragment order, zero-padded to
// Kpad = S*32.  packed[((cg*S + s)*32 + lane)*16 + i] = W[k(s,lane,i)][cg*16 + (lane&15)]
__global__ __launch_bounds__(256) void pack_w_kernel(const float* __restrict__ W,
                                                     _Float16* __restrict__ out,
                                                     int K, int S) {
  int cg = blockIdx.x / S, s = blockIdx.x % S;
  int tid = threadIdx.x;
  for (int idx = tid; idx < 512; idx += 256) {
    int lane = idx >> 4, i = idx & 15;
    int hf = lane >> 4, n = lane & 15;
    int k = s * 32 + ((i < 8) ? hf * 8 + i : 16 + hf * 8 + (i - 8));
    float v = (k < K) ? W[(size_t)k * 128 + cg * 16 + n] : 0.f;
    out[((size_t)(cg * S + s) * 32 + lane) * 16 + i] = (_Float16)v;
  }
}

__global__ void fill_kernel(float* __restrict__ p, float v, int n) {
  int i = blockIdx.x * 256 + threadIdx.x;
  if (i < n) p[i] = v;
}

__global__ void copy_kernel(const float* __restrict__ a, float* __restrict__ b, int n) {
  int i = blockIdx.x * 256 + threadIdx.x;
  if (i < n) b[i] = a[i];
}

__global__ void deg_count_kernel(const int* __restrict__ dst, float* __restrict__ deg, int n) {
  int i = blockIdx.x * 256 + threadIdx.x;
  if (i < n) atomicAdd(&deg[dst[i]], 1.0f);
}

__global__ void recip_kernel(const float* __restrict__ deg, float* __restrict__ invdeg, int n) {
  int i = blockIdx.x * 256 + threadIdx.x;
  if (i < n) invdeg[i] = 1.0f / (deg[i] + 1e-8f);
}

// ---------------------------------------------------------------------------
// Embedding: out = GELU(X@W1+b1)@W2+b2, Kin <= 32, 16-row strip per block.
__global__ __launch_bounds__(256) void embed_kernel(
    const float* __restrict__ X, int Kin,
    const _Float16* __restrict__ pW1, const float* __restrict__ b1,
    const _Float16* __restrict__ pW2, const float* __restrict__ b2,
    float* __restrict__ out) {
  __shared__ _Float16 ain[16 * EMB_LD];
  __shared__ _Float16 hid[16 * HID_LD];
  int tid = threadIdx.x;
  int r0 = blockIdx.x * 16;
  for (int idx = tid; idx < 16 * 32; idx += 256) {
    int r = idx >> 5, c = idx & 31;
    ain[r * EMB_LD + c] = (_Float16)((c < Kin) ? X[(size_t)(r0 + r) * Kin + c] : 0.f);
  }
  __syncthreads();
  int lane = tid & 31, w = tid >> 5, hf = lane >> 4, n = lane & 15;
  int col = w * 16 + n;
  v8f acc = ZACC;
  {
    v16h a = load_a(ain, EMB_LD, 0, lane);
    v16h bf = *(const v16h*)(pW1 + (size_t)w * 512 + lane * 16);
    acc = wmma16(a, bf, acc);
  }
  float bb = b1[col];
#pragma unroll
  for (int j = 0; j < 8; ++j)
    hid[(hf * 8 + j) * HID_LD + col] = (_Float16)gelu_f(acc[j] + bb);
  __syncthreads();
  v8f a2 = ZACC;
  for (int s = 0; s < 4; ++s) {
    v16h a = load_a(hid, HID_LD, s * 32, lane);
    v16h bf = *(const v16h*)(pW2 + (size_t)(w * 4 + s) * 512 + lane * 16);
    a2 = wmma16(a, bf, a2);
  }
  float b2v = b2[col];
#pragma unroll
  for (int j = 0; j < 8; ++j)
    out[(size_t)(r0 + hf * 8 + j) * HD + col] = a2[j] + b2v;
}

// ---------------------------------------------------------------------------
// Fused edge block, ESTR strips of 16 edges per workgroup.
// mi=[h[src],h[dst],e] per strip (shared by msg + edge_upd MLPs); each packed
// B-fragment load is reused across all ESTR A-tiles. msg -> atomic scatter to
// agg; edge_upd -> residual + LN -> e (in place). The fp32 pre-LN buffer
// aliases the mi tile (mi is dead after layer 1; barrier separates them).
__global__ __launch_bounds__(256) void edge_block_kernel(
    const float* __restrict__ h, float* __restrict__ e, float* __restrict__ agg,
    const int* __restrict__ src, const int* __restrict__ dst,
    const _Float16* __restrict__ pMsg1, const float* __restrict__ msgb1,
    const _Float16* __restrict__ pMsg2, const float* __restrict__ msgb2,
    const _Float16* __restrict__ pEdg1, const float* __restrict__ edgb1,
    const _Float16* __restrict__ pEdg2, const float* __restrict__ edgb2,
    const float* __restrict__ lng, const float* __restrict__ lnb) {
  __shared__ __align__(16) char mipre[ESTR * 16 * MI_LD * 2]; // f16 mi | f32 pre
  _Float16* mi = (_Float16*)mipre;
  float* pre = (float*)mipre;                                 // 64*PRE_LD*4 <= sizeof(mipre)
  __shared__ _Float16 hidm[ESTR * 16 * HID_LD];
  __shared__ _Float16 hide[ESTR * 16 * HID_LD];
  __shared__ float mu[ESTR * 16], rs[ESTR * 16];
  __shared__ int rsrc[ESTR * 16], rdst[ESTR * 16];

  int tid = threadIdx.x;
  int e0 = blockIdx.x * (16 * ESTR);
  if (tid < 16 * ESTR) { rsrc[tid] = src[e0 + tid]; rdst[tid] = dst[e0 + tid]; }
  __syncthreads();
  for (int idx = tid; idx < 16 * ESTR * 384; idx += 256) {
    int r = idx / 384, c = idx - r * 384;
    float v;
    if (c < 128)      v = h[(size_t)rsrc[r] * HD + c];
    else if (c < 256) v = h[(size_t)rdst[r] * HD + (c - 128)];
    else              v = e[(size_t)(e0 + r) * HD + (c - 256)];
    mi[r * MI_LD + c] = (_Float16)v;
  }
  __syncthreads();

  int lane = tid & 31, w = tid >> 5, hf = lane >> 4, n = lane & 15;
  int col = w * 16 + n;

  {
    v8f am[ESTR], ae[ESTR];
    v8f z = ZACC;
#pragma unroll
    for (int t = 0; t < ESTR; ++t) { am[t] = z; ae[t] = z; }
    const _Float16* pm1 = pMsg1 + (size_t)(w * 12) * 512;
    const _Float16* pe1 = pEdg1 + (size_t)(w * 12) * 512;
    for (int s = 0; s < 12; ++s) {
      v16h bm = *(const v16h*)(pm1 + (size_t)s * 512 + lane * 16);
      v16h be = *(const v16h*)(pe1 + (size_t)s * 512 + lane * 16);
#pragma unroll
      for (int t = 0; t < ESTR; ++t) {
        v16h a = load_a(mi + t * 16 * MI_LD, MI_LD, s * 32, lane);
        am[t] = wmma16(a, bm, am[t]);
        ae[t] = wmma16(a, be, ae[t]);
      }
    }
    float bm1 = msgb1[col], be1 = edgb1[col];
#pragma unroll
    for (int t = 0; t < ESTR; ++t) {
#pragma unroll
      for (int j = 0; j < 8; ++j) {
        int m = t * 16 + hf * 8 + j;
        hidm[m * HID_LD + col] = (_Float16)gelu_f(am[t][j] + bm1);
        hide[m * HID_LD + col] = (_Float16)gelu_f(ae[t][j] + be1);
      }
    }
  }
  __syncthreads();   // hidden ready; mi dead -> pre may now be written

  {
    v8f am2[ESTR], ae2[ESTR];
    v8f z = ZACC;
#pragma unroll
    for (int t = 0; t < ESTR; ++t) { am2[t] = z; ae2[t] = z; }
    const _Float16* pm2 = pMsg2 + (size_t)(w * 4) * 512;
    const _Float16* pe2 = pEdg2 + (size_t)(w * 4) * 512;
    for (int s = 0; s < 4; ++s) {
      v16h bm = *(const v16h*)(pm2 + (size_t)s * 512 + lane * 16);
      v16h be = *(const v16h*)(pe2 + (size_t)s * 512 + lane * 16);
#pragma unroll
      for (int t = 0; t < ESTR; ++t) {
        v16h a1 = load_a(hidm + t * 16 * HID_LD, HID_LD, s * 32, lane);
        am2[t] = wmma16(a1, bm, am2[t]);
        v16h a2 = load_a(hide + t * 16 * HID_LD, HID_LD, s * 32, lane);
        ae2[t] = wmma16(a2, be, ae2[t]);
      }
    }
    float bm2 = msgb2[col], be2 = edgb2[col];
#pragma unroll
    for (int t = 0; t < ESTR; ++t) {
#pragma unroll
      for (int j = 0; j < 8; ++j) {
        int m = t * 16 + hf * 8 + j;
        atomicAdd(&agg[(size_t)rdst[m] * HD + col], am2[t][j] + bm2);
        pre[m * PRE_LD + col] = e[(size_t)(e0 + m) * HD + col] + ae2[t][j] + be2;
      }
    }
  }
  __syncthreads();
  if (tid < 16 * ESTR) {
    float s1 = 0.f;
    for (int c = 0; c < 128; ++c) s1 += pre[tid * PRE_LD + c];
    float mv = s1 * (1.f / 128.f);
    float s2 = 0.f;
    for (int c = 0; c < 128; ++c) { float d = pre[tid * PRE_LD + c] - mv; s2 += d * d; }
    mu[tid] = mv;
    rs[tid] = rsqrtf(s2 * (1.f / 128.f) + 1e-5f);
  }
  __syncthreads();
  for (int idx = tid; idx < 16 * ESTR * 128; idx += 256) {
    int r = idx >> 7, c = idx & 127;
    e[(size_t)(e0 + r) * HD + c] = (pre[r * PRE_LD + c] - mu[r]) * rs[r] * lng[c] + lnb[c];
  }
}

// ---------------------------------------------------------------------------
// Node update: in=[h, agg*invdeg] (16x256) -> MLP -> residual + LN -> hout
__global__ __launch_bounds__(256) void node_block_kernel(
    const float* __restrict__ h, const float* __restrict__ agg,
    const float* __restrict__ invdeg,
    const _Float16* __restrict__ pW1, const float* __restrict__ b1,
    const _Float16* __restrict__ pW2, const float* __restrict__ b2,
    const float* __restrict__ lng, const float* __restrict__ lnb,
    float* __restrict__ hout) {
  __shared__ _Float16 ain[16 * AIN_LD];
  __shared__ _Float16 hid[16 * HID_LD];
  __shared__ float pre[16 * PRE_LD];
  __shared__ float mu[16], rs[16];
  int tid = threadIdx.x;
  int r0 = blockIdx.x * 16;
  for (int idx = tid; idx < 16 * 256; idx += 256) {
    int r = idx >> 8, c = idx & 255;
    int node = r0 + r;
    float v = (c < 128) ? h[(size_t)node * HD + c]
                        : agg[(size_t)node * HD + (c - 128)] * invdeg[node];
    ain[r * AIN_LD + c] = (_Float16)v;
  }
  __syncthreads();
  int lane = tid & 31, w = tid >> 5, hf = lane >> 4, n = lane & 15;
  int col = w * 16 + n;
  v8f acc = ZACC;
  for (int s = 0; s < 8; ++s) {
    v16h a = load_a(ain, AIN_LD, s * 32, lane);
    v16h bf = *(const v16h*)(pW1 + (size_t)(w * 8 + s) * 512 + lane * 16);
    acc = wmma16(a, bf, acc);
  }
  float bb = b1[col];
#pragma unroll
  for (int j = 0; j < 8; ++j)
    hid[(hf * 8 + j) * HID_LD + col] = (_Float16)gelu_f(acc[j] + bb);
  __syncthreads();
  v8f a2 = ZACC;
  for (int s = 0; s < 4; ++s) {
    v16h a = load_a(hid, HID_LD, s * 32, lane);
    v16h bf = *(const v16h*)(pW2 + (size_t)(w * 4 + s) * 512 + lane * 16);
    a2 = wmma16(a, bf, a2);
  }
  float b2v = b2[col];
#pragma unroll
  for (int j = 0; j < 8; ++j) {
    int m = hf * 8 + j;
    pre[m * PRE_LD + col] = h[(size_t)(r0 + m) * HD + col] + a2[j] + b2v;
  }
  __syncthreads();
  if (tid < 16) {
    float s1 = 0.f;
    for (int c = 0; c < 128; ++c) s1 += pre[tid * PRE_LD + c];
    float mv = s1 * (1.f / 128.f);
    float s2 = 0.f;
    for (int c = 0; c < 128; ++c) { float d = pre[tid * PRE_LD + c] - mv; s2 += d * d; }
    mu[tid] = mv;
    rs[tid] = rsqrtf(s2 * (1.f / 128.f) + 1e-5f);
  }
  __syncthreads();
  for (int idx = tid; idx < 16 * 128; idx += 256) {
    int r = idx >> 7, c = idx & 127;
    hout[(size_t)(r0 + r) * HD + c] = (pre[r * PRE_LD + c] - mu[r]) * rs[r] * lng[c] + lnb[c];
  }
}

// ---------------------------------------------------------------------------
// Column-wise mean/max over all N rows -> g[256] (and mirrored to d_out)
__global__ __launch_bounds__(256) void readout_kernel(const float* __restrict__ h,
                                                      float* __restrict__ g,
                                                      float* __restrict__ gout) {
  int c = blockIdx.x;
  int tid = threadIdx.x;
  __shared__ float ssum[256], smax[256];
  float s = 0.f, mx = -3.402823466e38f;
  for (int r = tid; r < NN; r += 256) {
    float v = h[(size_t)r * HD + c];
    s += v;
    mx = fmaxf(mx, v);
  }
  ssum[tid] = s; smax[tid] = mx;
  __syncthreads();
  for (int o = 128; o > 0; o >>= 1) {
    if (tid < o) {
      ssum[tid] += ssum[tid + o];
      smax[tid] = fmaxf(smax[tid], smax[tid + o]);
    }
    __syncthreads();
  }
  if (tid == 0) {
    float mean = ssum[0] * (1.f / (float)NN);
    g[c] = mean;         g[128 + c] = smax[0];
    gout[c] = mean;      gout[128 + c] = smax[0];
  }
}

// One small head: out = act((GELU(g@W1+b1))@W2+b2). act: 0=id,1=sigmoid,2=softplus
__global__ __launch_bounds__(128) void head_kernel(
    const float* __restrict__ g,
    const float* __restrict__ W1, const float* __restrict__ b1,
    const float* __restrict__ W2, const float* __restrict__ b2,
    int outdim, int act, float* __restrict__ out) {
  __shared__ float hid[128];
  int tid = threadIdx.x;
  float a = b1[tid];
  for (int k = 0; k < 256; ++k) a += g[k] * W1[k * 128 + tid];
  hid[tid] = gelu_f(a);
  __syncthreads();
  if (tid < outdim) {
    float o = b2[tid];
    for (int j = 0; j < 128; ++j) o += hid[j] * W2[j * outdim + tid];
    if (act == 1)      o = 1.f / (1.f + expf(-o));
    else if (act == 2) o = (o > 20.f) ? o : log1pf(expf(o));
    out[tid] = o;
  }
}

// ---------------------------------------------------------------------------
extern "C" void kernel_launch(void* const* d_in, const int* in_sizes, int n_in,
                              void* d_out, int out_size, void* d_ws, size_t ws_size,
                              hipStream_t stream) {
  (void)in_sizes; (void)n_in; (void)out_size; (void)ws_size;
  const float* node_raw = (const float*)d_in[0];
  const float* edge_raw = (const float*)d_in[1];
  const int* eidx = (const int*)d_in[2];
  const int* srcp = eidx;
  const int* dstp = eidx + EE;
  auto F = [&](int i) { return (const float*)d_in[i]; };

  char* ws = (char*)d_ws;
  size_t off = 0;
  auto alloc = [&](size_t bytes) -> void* {
    void* p = ws + off;
    off += (bytes + 255) & ~(size_t)255;
    return p;
  };
  float* h0 = (float*)alloc(sizeof(float) * (size_t)NN * HD);
  float* h1 = (float*)alloc(sizeof(float) * (size_t)NN * HD);
  float* e  = (float*)alloc(sizeof(float) * (size_t)EE * HD);
  float* agg = (float*)alloc(sizeof(float) * (size_t)NN * HD);
  float* deg = (float*)alloc(sizeof(float) * (size_t)NN);
  float* invdeg = (float*)alloc(sizeof(float) * (size_t)NN);
  float* gbuf = (float*)alloc(sizeof(float) * 256);

  auto packW = [&](const float* W, int K) -> _Float16* {
    int S = (K + 31) / 32;
    _Float16* p = (_Float16*)alloc(sizeof(_Float16) * (size_t)S * 4096);
    pack_w_kernel<<<dim3(8 * S), dim3(256), 0, stream>>>(W, p, K, S);
    return p;
  };

  _Float16* pNE1 = packW(F(3), 18);
  _Float16* pNE2 = packW(F(5), 128);
  _Float16* pEE1 = packW(F(7), 11);
  _Float16* pEE2 = packW(F(9), 128);
  _Float16 *pMsg1[6], *pMsg2[6], *pNod1[6], *pNod2[6], *pEdg1[6], *pEdg2[6];
  for (int b = 0; b < 6; ++b) {
    int base = 11 + 16 * b;
    pMsg1[b] = packW(F(base + 0), 384);
    pMsg2[b] = packW(F(base + 2), 128);
    pNod1[b] = packW(F(base + 4), 256);
    pNod2[b] = packW(F(base + 6), 128);
    pEdg1[b] = packW(F(base + 8), 384);
    pEdg2[b] = packW(F(base + 10), 128);
  }

  // Embeddings
  embed_kernel<<<NN / 16, 256, 0, stream>>>(node_raw, 18, pNE1, F(4), pNE2, F(6), h0);
  embed_kernel<<<EE / 16, 256, 0, stream>>>(edge_raw, 11, pEE1, F(8), pEE2, F(10), e);

  // Degree (constant across blocks) and its reciprocal
  fill_kernel<<<(NN + 255) / 256, 256, 0, stream>>>(deg, 0.f, NN);
  deg_count_kernel<<<(EE + 255) / 256, 256, 0, stream>>>(dstp, deg, EE);
  recip_kernel<<<(NN + 255) / 256, 256, 0, stream>>>(deg, invdeg, NN);

  float* hc = h0;
  float* hn = h1;
  for (int b = 0; b < 6; ++b) {
    int base = 11 + 16 * b;
    fill_kernel<<<(NN * HD + 255) / 256, 256, 0, stream>>>(agg, 0.f, NN * HD);
    edge_block_kernel<<<EE / (16 * ESTR), 256, 0, stream>>>(
        hc, e, agg, srcp, dstp,
        pMsg1[b], F(base + 1), pMsg2[b], F(base + 3),
        pEdg1[b], F(base + 9), pEdg2[b], F(base + 11),
        F(base + 14), F(base + 15));
    node_block_kernel<<<NN / 16, 256, 0, stream>>>(
        hc, agg, invdeg,
        pNod1[b], F(base + 5), pNod2[b], F(base + 7),
        F(base + 12), F(base + 13), hn);
    float* t = hc; hc = hn; hn = t;
  }

  float* out = (float*)d_out;
  // outputs: fe(1) st(1) hl(1) coord(4) solv(3) h(N*H) g(256)
  copy_kernel<<<(NN * HD + 255) / 256, 256, 0, stream>>>(hc, out + 10, NN * HD);
  readout_kernel<<<128, 256, 0, stream>>>(hc, gbuf, out + 10 + (size_t)NN * HD);

  int hb = 107;
  head_kernel<<<1, 128, 0, stream>>>(gbuf, F(hb + 0), F(hb + 1), F(hb + 2), F(hb + 3), 1, 0, out + 0);
  head_kernel<<<1, 128, 0, stream>>>(gbuf, F(hb + 4), F(hb + 5), F(hb + 6), F(hb + 7), 1, 1, out + 1);
  head_kernel<<<1, 128, 0, stream>>>(gbuf, F(hb + 8), F(hb + 9), F(hb + 10), F(hb + 11), 1, 2, out + 2);
  head_kernel<<<1, 128, 0, stream>>>(gbuf, F(hb + 12), F(hb + 13), F(hb + 14), F(hb + 15), 4, 0, out + 3);
  head_kernel<<<1, 128, 0, stream>>>(gbuf, F(hb + 16), F(hb + 17), F(hb + 18), F(hb + 19), 3, 0, out + 7);
}
